// GNNInferenceNetwork_86638080295570
// MI455X (gfx1250) — matrix operand order repaired and checked by reading, not compiled, not run on hardware
//
#include <hip/hip_runtime.h>
#include <hip/hip_fp16.h>

// GNN inference, MI455X (gfx1250, wave32).
// Heavy MLPs run on v_wmma_f32_16x16x32_f16 (f32 accumulate); weights are
// prepacked into B-fragment lane order once per launch; edge-kernel A panels
// are staged into LDS via TDM (tensor_load_to_lds) + async global->LDS loads;
// the final (output-producing) globals MLP runs in fp32 VALU.

typedef __attribute__((ext_vector_type(16))) _Float16 v16h;
typedef __attribute__((ext_vector_type(8)))  float    v8f;
typedef __attribute__((ext_vector_type(4)))  unsigned int v4u;
typedef __attribute__((ext_vector_type(8)))  int      v8i_;
typedef __attribute__((ext_vector_type(4)))  int      v4i_;

#define GNN_B 512
#define GNN_S 32
#define GNN_D 128
#define GNN_R 64

#define WMMA_F16(a, b, c) \
  __builtin_amdgcn_wmma_f32_16x16x32_f16(false, (a), false, (b), (short)0, (c), false, false)

// ---- CDNA5 async data movement helpers --------------------------------------

// Per-lane async 16B copy global->LDS (ASYNCcnt). Generic LDS pointer low 32
// bits are the workgroup-relative LDS offset (aperture in the high bits).
static __device__ __forceinline__ void async_b128_to_lds(unsigned lds_off,
                                                         const void* gsrc) {
  asm volatile("global_load_async_to_lds_b128 %0, %1, off"
               :
               : "v"(lds_off), "v"((unsigned long long)(uintptr_t)gsrc)
               : "memory");
}
static __device__ __forceinline__ void wait_asynccnt0() {
  asm volatile("s_wait_asynccnt 0x0" ::: "memory");
}

// ---- fragment helpers -------------------------------------------------------

// A fragment (16x32, f16) from LDS, per ISA "16-bit A-Matrix 16x32" layout:
// lane<16: rows 0..15, K halves {0..7,16..23}; lane>=16: K halves {8..15,24..31}.
// Per lane this is two contiguous 16B chunks -> compiler emits ds_load_b128 x2.
static __device__ __forceinline__ v16h frag_a_lds(const _Float16* As, int lda,
                                                  int mbase, int k0) {
  const int lane = threadIdx.x & 31;
  const int half = lane >> 4;
  const _Float16* rp = As + (mbase + (lane & 15)) * lda + k0 + 8 * half;
  v16h a;
#pragma unroll
  for (int i = 0; i < 8; ++i) {
    const int k = (i < 4) ? (2 * i) : (8 + 2 * i);  // 0,2,4,6,16,18,20,22
    a[2 * i]     = rp[k];
    a[2 * i + 1] = rp[k + 1];
  }
  return a;
}

// B fragment from the prepacked weight buffer: one contiguous v16h per lane.
static __device__ __forceinline__ v16h frag_b_packed(const _Float16* pw, int nkt,
                                                     int nt, int kt) {
  const v16h* p = (const v16h*)pw;
  return p[((size_t)(nt * nkt + kt)) * 32 + (threadIdx.x & 31)];
}

// C/D (16x16 f32) layout: VGPR r -> row r+8*half, col = lane&15.
static __device__ __forceinline__ void store_tile_f16(v8f c, _Float16* Hs, int ldh,
                                                      int mbase, int nbase) {
  const int lane = threadIdx.x & 31;
  const int half = lane >> 4;
  const int col = nbase + (lane & 15);
#pragma unroll
  for (int r = 0; r < 8; ++r)
    Hs[(mbase + 8 * half + r) * ldh + col] = (_Float16)c[r];
}

// ---- weight prepack: W (KxN f32, row-major) -> B-fragment order (f16) -------
// Fragment (nt,kt), lane: 16 halves. lane<16: K=kt*32+{0..7,16..23}; lane>=16: +8.
__global__ __launch_bounds__(256) void pack_weight_kernel(const float* W,
                                                          _Float16* out, int K,
                                                          int N) {
  const int idx = blockIdx.x * 256 + threadIdx.x;
  const int nkt = K >> 5;
  const int total = (N >> 4) * nkt * 32;
  if (idx >= total) return;
  const int lane = idx & 31;
  const int frag = idx >> 5;
  const int kt = frag % nkt;
  const int ntile = frag / nkt;
  const int half = lane >> 4;
  const int col = ntile * 16 + (lane & 15);
  _Float16* o = out + (size_t)idx * 16;
#pragma unroll
  for (int i = 0; i < 8; ++i) {
    const int k = kt * 32 + 8 * half + ((i < 4) ? (2 * i) : (8 + 2 * i));
    o[2 * i]     = (_Float16)W[(size_t)k * N + col];
    o[2 * i + 1] = (_Float16)W[(size_t)(k + 1) * N + col];
  }
}

__global__ __launch_bounds__(256) void zero_kernel(float* p, int n) {
  const int i = blockIdx.x * 256 + threadIdx.x;
  if (i < n) p[i] = 0.f;
}

// ---- node MLP2: per-batch 32xKIN @ KINx128 -> relu -> @128x128 -> relu ------
template <int KIN, bool CONCAT_MSGS, bool OUT_SUM>
__global__ __launch_bounds__(512) void node_mlp_kernel(
    const float* x_f32, const _Float16* ne_prev, const float* msgs,
    const _Float16* pw1, const float* b1, const _Float16* pw2, const float* b2,
    _Float16* ne_out, float* nsum) {
  constexpr int LDA = KIN + 8;
  constexpr int LDH = 128 + 8;
  __shared__ _Float16 As[32 * LDA];
  __shared__ _Float16 Hs[32 * LDH];
  __shared__ float ns_s[128];
  const int b = blockIdx.x;
  const int tid = threadIdx.x;
  if constexpr (OUT_SUM) {
    if (tid < 128) ns_s[tid] = 0.f;
  }
  for (int idx = tid; idx < 32 * KIN; idx += 512) {
    const int row = idx / KIN, col = idx - row * KIN;
    float v;
    if constexpr (!CONCAT_MSGS) {
      v = x_f32[((size_t)b * 32 + row) * 128 + col];
    } else {
      v = (col < 128) ? (float)ne_prev[((size_t)b * 32 + row) * 128 + col]
                      : msgs[((size_t)b * 32 + row) * 64 + (col - 128)];
    }
    As[row * LDA + col] = (_Float16)v;
  }
  __syncthreads();
  const int wave = tid >> 5, lane = tid & 31;
  const int mt = wave >> 3, nt = wave & 7;  // 2 x 8 tiles of 32x128
  constexpr int NKT1 = KIN / 32;
  v8f c = {};
#pragma unroll
  for (int kt = 0; kt < NKT1; ++kt)
    c = WMMA_F16(frag_a_lds(As, LDA, mt * 16, kt * 32),
                 frag_b_packed(pw1, NKT1, nt, kt), c);
  {
    const float bias = b1[nt * 16 + (lane & 15)];
#pragma unroll
    for (int r = 0; r < 8; ++r) c[r] = fmaxf(c[r] + bias, 0.f);
  }
  store_tile_f16(c, Hs, LDH, mt * 16, nt * 16);
  __syncthreads();
  v8f d = {};
#pragma unroll
  for (int kt = 0; kt < 4; ++kt)
    d = WMMA_F16(frag_a_lds(Hs, LDH, mt * 16, kt * 32),
                 frag_b_packed(pw2, 4, nt, kt), d);
  const float bias2 = b2[nt * 16 + (lane & 15)];
  const int col = nt * 16 + (lane & 15);
  const int half = lane >> 4;
  float csum = 0.f;
#pragma unroll
  for (int r = 0; r < 8; ++r) {
    const float v = fmaxf(d[r] + bias2, 0.f);
    csum += v;
    ne_out[((size_t)b * 32 + mt * 16 + 8 * half + r) * 128 + col] = (_Float16)v;
  }
  if constexpr (OUT_SUM) {
    atomicAdd(&ns_s[col], csum);
    __syncthreads();
    if (tid < 128) nsum[(size_t)b * 128 + tid] = ns_s[tid];
  }
}

// ---- edge encoder: per (b,i), E = concat(x_i, x_j) (32x256) -> em MLP -------
__global__ __launch_bounds__(256) void edge_encode_kernel(
    const float* x, const _Float16* pw1, const float* b1, const _Float16* pw2,
    const float* b2, _Float16* ee_out, float* msgs_out) {
  constexpr int KIN = 256, LDA = KIN + 8, LDH = 64 + 8;
  __shared__ _Float16 As[32 * LDA];
  __shared__ _Float16 Hs[32 * LDH];
  __shared__ float ms[64];
  const int bi = blockIdx.x;
  const int b = bi >> 5, i = bi & 31;
  const int tid = threadIdx.x;
  if (tid < 64) ms[tid] = 0.f;
  const float* xb = x + (size_t)b * 32 * 128;
  for (int idx = tid; idx < 32 * KIN; idx += 256) {
    const int row = idx >> 8, col = idx & 255;
    const float v = (col < 128) ? xb[i * 128 + col] : xb[row * 128 + (col - 128)];
    As[row * LDA + col] = (_Float16)v;
  }
  __syncthreads();
  const int wave = tid >> 5, lane = tid & 31;
  const int mt = wave >> 2, nt = wave & 3;  // 2 x 4 tiles of 32x64
  v8f c = {};
#pragma unroll
  for (int kt = 0; kt < 8; ++kt)
    c = WMMA_F16(frag_a_lds(As, LDA, mt * 16, kt * 32),
                 frag_b_packed(pw1, 8, nt, kt), c);
  {
    const float bias = b1[nt * 16 + (lane & 15)];
#pragma unroll
    for (int r = 0; r < 8; ++r) c[r] = fmaxf(c[r] + bias, 0.f);
  }
  store_tile_f16(c, Hs, LDH, mt * 16, nt * 16);
  __syncthreads();
  v8f d = {};
#pragma unroll
  for (int kt = 0; kt < 2; ++kt)
    d = WMMA_F16(frag_a_lds(Hs, LDH, mt * 16, kt * 32),
                 frag_b_packed(pw2, 2, nt, kt), d);
  const float bias2 = b2[nt * 16 + (lane & 15)];
  const int col = nt * 16 + (lane & 15);
  const int half = lane >> 4;
  float csum = 0.f;
  _Float16* erow = ee_out + (size_t)bi * 32 * 64;
#pragma unroll
  for (int r = 0; r < 8; ++r) {
    const float v = fmaxf(d[r] + bias2, 0.f);
    csum += v;
    erow[(mt * 16 + 8 * half + r) * 64 + col] = (_Float16)v;
  }
  atomicAdd(&ms[col], csum);
  __syncthreads();
  if (tid < 64) msgs_out[(size_t)bi * 64 + tid] = ms[tid] * (1.f / 32.f);
}

// ---- edge update: per (b,i), E = concat(ee, ne_i, ne_j) (32x320) -> eu MLP --
// A panel split: EEs (32x64, TDM tensor_load_to_lds) | Ns (32x256, async->LDS).
template <bool WRITE_EE>
__global__ __launch_bounds__(256) void edge_update_kernel(
    const _Float16* ee_in, const _Float16* ne, const _Float16* pw1,
    const float* b1, const _Float16* pw2, const float* b2, _Float16* ee_out,
    float* msgs_out, float* esum) {
  constexpr int LDE = 64 + 8;    // EE tile row stride (halves)
  constexpr int LDN = 256 + 8;   // N panel row stride (halves)
  constexpr int LDH = 64 + 8;
  __shared__ _Float16 EEs[32 * LDE];
  __shared__ _Float16 Ns[32 * LDN];
  __shared__ _Float16 Hs[32 * LDH];
  __shared__ float ms[64];
  const int bi = blockIdx.x;
  const int b = bi >> 5, i = bi & 31;
  const int tid = threadIdx.x;
  if (tid < 64) ms[tid] = 0.f;
  const _Float16* ein = ee_in + (size_t)bi * 32 * 64;
  const _Float16* nb = ne + (size_t)b * 32 * 128;

  // --- EE tile (32x64 f16, 128B rows) via Tensor Data Mover, wave 0 only ---
#if __has_builtin(__builtin_amdgcn_tensor_load_to_lds)
  if (tid < 32) {
    const unsigned lds_ee = (unsigned)(uintptr_t)&EEs[0];  // low 32 = LDS offset
    const unsigned long long ga = (unsigned long long)(uintptr_t)ein;
    // D# group0: count=1(valid); lds_addr; 57-bit global addr; type=2.
    const v4u g0 = {1u, lds_ee, (unsigned)ga,
                    (unsigned)((ga >> 32) & 0x1FFFFFFu) | 0x80000000u};
    // D# group1: data_size=2B, pad_enable, pad_interval=32dw, pad_amount=4dw;
    // tensor_dim0=64, tensor_dim1=32, tile_dim0=64, tile_dim1=32,
    // tensor_dim0_stride=64, tensor_dim1_stride=2048.
    const v8i_ g1 = {(int)0x07110000, (int)(64u << 16), (int)(32u << 16),
                     (int)(64u << 16), 32, 64, (int)0x08000000, 0};
    const v4i_ gz = {0, 0, 0, 0};
#if __clang_major__ >= 23
    const v8i_ gz8 = {0, 0, 0, 0, 0, 0, 0, 0};
    __builtin_amdgcn_tensor_load_to_lds(g0, g1, gz, gz, gz8, 0);
#else
    __builtin_amdgcn_tensor_load_to_lds(g0, g1, gz, gz, 0);
#endif
    __builtin_amdgcn_s_wait_tensorcnt(0);
  }
#else
  for (int idx = tid; idx < 32 * 64; idx += 256)
    EEs[(idx >> 6) * LDE + (idx & 63)] = ein[idx];
#endif

  // --- N panel (32 x [ne_i | ne_j]) via per-lane async global->LDS b128 ---
  {
    const unsigned ns_base = (unsigned)(uintptr_t)&Ns[0];
#pragma unroll
    for (int k = 0; k < 4; ++k) {
      const int c = tid + k * 256;       // 1024 chunks of 8 halves (16B)
      const int row = c >> 5;
      const int col = (c & 31) * 8;
      const _Float16* src =
          (col < 128) ? (nb + i * 128 + col) : (nb + row * 128 + (col - 128));
      async_b128_to_lds(ns_base + (unsigned)(row * LDN + col) * 2, src);
    }
    wait_asynccnt0();
  }
  __syncthreads();

  const int wave = tid >> 5, lane = tid & 31;
  const int mt = wave >> 2, nt = wave & 3;  // 2 x 4 tiles of 32x64
  v8f c = {};
#pragma unroll
  for (int kt = 0; kt < 2; ++kt)  // K 0..63 from EE tile
    c = WMMA_F16(frag_a_lds(EEs, LDE, mt * 16, kt * 32),
                 frag_b_packed(pw1, 10, nt, kt), c);
#pragma unroll
  for (int kt = 2; kt < 10; ++kt)  // K 64..319 from N panel
    c = WMMA_F16(frag_a_lds(Ns, LDN, mt * 16, (kt - 2) * 32),
                 frag_b_packed(pw1, 10, nt, kt), c);
  {
    const float bias = b1[nt * 16 + (lane & 15)];
#pragma unroll
    for (int r = 0; r < 8; ++r) c[r] = fmaxf(c[r] + bias, 0.f);
  }
  store_tile_f16(c, Hs, LDH, mt * 16, nt * 16);
  __syncthreads();
  v8f d = {};
#pragma unroll
  for (int kt = 0; kt < 2; ++kt)
    d = WMMA_F16(frag_a_lds(Hs, LDH, mt * 16, kt * 32),
                 frag_b_packed(pw2, 2, nt, kt), d);
  const float bias2 = b2[nt * 16 + (lane & 15)];
  const int col = nt * 16 + (lane & 15);
  const int half = lane >> 4;
  float csum = 0.f;
#pragma unroll
  for (int r = 0; r < 8; ++r) {
    const float v = fmaxf(d[r] + bias2, 0.f);
    csum += v;
    if constexpr (WRITE_EE)
      ee_out[(size_t)bi * 32 * 64 + (mt * 16 + 8 * half + r) * 64 + col] =
          (_Float16)v;
  }
  atomicAdd(&ms[col], csum);
  __syncthreads();
  if constexpr (WRITE_EE) {
    if (tid < 64) msgs_out[(size_t)bi * 64 + tid] = ms[tid] * (1.f / 32.f);
  } else {
    if (tid < 64) atomicAdd(&esum[(size_t)b * 64 + tid], ms[tid]);
  }
}

// ---- globals MLP (fp32 VALU; tiny) + softplus, writes d_out -----------------
__global__ __launch_bounds__(256) void globals_kernel(
    const float* nsum, const float* esum, const float* w1, const float* bb1,
    const float* w2, const float* bb2, const float* w3, const float* bb3,
    float* out) {
  __shared__ float gin[192];
  __shared__ float h1[256];
  __shared__ float h2[256];
  const int b = blockIdx.x;
  const int t = threadIdx.x;
  if (t < 128) gin[t] = nsum[(size_t)b * 128 + t] * (1.f / 32.f);
  else if (t < 192) gin[t] = esum[(size_t)b * 64 + (t - 128)] * (1.f / 1024.f);
  __syncthreads();
  {
    float acc = bb1[t];
    for (int k = 0; k < 192; ++k) acc = fmaf(gin[k], w1[k * 256 + t], acc);
    h1[t] = fmaxf(acc, 0.f);
  }
  __syncthreads();
  {
    float acc = bb2[t];
    for (int k = 0; k < 256; ++k) acc = fmaf(h1[k], w2[k * 256 + t], acc);
    h2[t] = fmaxf(acc, 0.f);
  }
  __syncthreads();
  if (t < 128) {
    float acc = bb3[t];
    for (int k = 0; k < 256; ++k) acc = fmaf(h2[k], w3[k * 128 + t], acc);
    if (t < 64) {
      out[(size_t)b * 64 + t] = acc;  // mu
    } else {
      const float sp = (acc > 20.f) ? acc : log1pf(expf(acc));  // softplus
      out[(size_t)GNN_B * GNN_R + (size_t)b * 64 + (t - 64)] = sp;
    }
  }
}

// ---- launch -----------------------------------------------------------------
extern "C" void kernel_launch(void* const* d_in, const int* in_sizes, int n_in,
                              void* d_out, int out_size, void* d_ws,
                              size_t ws_size, hipStream_t stream) {
  (void)in_sizes; (void)n_in; (void)out_size; (void)ws_size;
  const float* x     = (const float*)d_in[0];
  const float* nm_w1 = (const float*)d_in[1];
  const float* nm_b1 = (const float*)d_in[2];
  const float* nm_w2 = (const float*)d_in[3];
  const float* nm_b2 = (const float*)d_in[4];
  const float* em_w1 = (const float*)d_in[5];
  const float* em_b1 = (const float*)d_in[6];
  const float* em_w2 = (const float*)d_in[7];
  const float* em_b2 = (const float*)d_in[8];
  const float* nu_w1 = (const float*)d_in[9];
  const float* nu_b1 = (const float*)d_in[10];
  const float* nu_w2 = (const float*)d_in[11];
  const float* nu_b2 = (const float*)d_in[12];
  const float* eu_w1 = (const float*)d_in[13];
  const float* eu_b1 = (const float*)d_in[14];
  const float* eu_w2 = (const float*)d_in[15];
  const float* eu_b2 = (const float*)d_in[16];
  const float* gm_w1 = (const float*)d_in[17];
  const float* gm_b1 = (const float*)d_in[18];
  const float* gm_w2 = (const float*)d_in[19];
  const float* gm_b2 = (const float*)d_in[20];
  const float* gm_w3 = (const float*)d_in[21];
  const float* gm_b3 = (const float*)d_in[22];

  unsigned char* ws = (unsigned char*)d_ws;
  size_t off = 0;
  auto take = [&](size_t bytes) -> void* {
    off = (off + 255) & ~(size_t)255;
    void* p = ws + off;
    off += bytes;
    return p;
  };
  _Float16* pw_nm1 = (_Float16*)take((size_t)128 * 128 * 2);
  _Float16* pw_nm2 = (_Float16*)take((size_t)128 * 128 * 2);
  _Float16* pw_em1 = (_Float16*)take((size_t)256 * 64 * 2);
  _Float16* pw_em2 = (_Float16*)take((size_t)64 * 64 * 2);
  _Float16* pw_nu1 = (_Float16*)take((size_t)192 * 128 * 2);
  _Float16* pw_nu2 = (_Float16*)take((size_t)128 * 128 * 2);
  _Float16* pw_eu1 = (_Float16*)take((size_t)320 * 64 * 2);
  _Float16* pw_eu2 = (_Float16*)take((size_t)64 * 64 * 2);
  _Float16* ne0   = (_Float16*)take((size_t)GNN_B * 32 * 128 * 2);
  _Float16* ne1   = (_Float16*)take((size_t)GNN_B * 32 * 128 * 2);
  _Float16* ne2   = (_Float16*)take((size_t)GNN_B * 32 * 128 * 2);
  float*    msgs1 = (float*)take((size_t)GNN_B * 32 * 64 * 4);
  float*    msgs2 = (float*)take((size_t)GNN_B * 32 * 64 * 4);
  _Float16* ee1   = (_Float16*)take((size_t)GNN_B * 32 * 32 * 64 * 2);
  _Float16* ee2   = (_Float16*)take((size_t)GNN_B * 32 * 32 * 64 * 2);
  float*    nsum  = (float*)take((size_t)GNN_B * 128 * 4);
  float*    esum  = (float*)take((size_t)GNN_B * 64 * 4);

  auto pack = [&](const float* W, _Float16* o, int K, int N) {
    const int total = (N / 16) * (K / 32) * 32;
    pack_weight_kernel<<<(total + 255) / 256, 256, 0, stream>>>(W, o, K, N);
  };
  pack(nm_w1, pw_nm1, 128, 128);
  pack(nm_w2, pw_nm2, 128, 128);
  pack(em_w1, pw_em1, 256, 64);
  pack(em_w2, pw_em2, 64, 64);
  pack(nu_w1, pw_nu1, 192, 128);
  pack(nu_w2, pw_nu2, 128, 128);
  pack(eu_w1, pw_eu1, 320, 64);
  pack(eu_w2, pw_eu2, 64, 64);

  zero_kernel<<<(GNN_B * 64 + 255) / 256, 256, 0, stream>>>(esum, GNN_B * 64);

  // ne0 = MLP2(x, nm)
  node_mlp_kernel<128, false, false><<<GNN_B, 512, 0, stream>>>(
      x, nullptr, nullptr, pw_nm1, nm_b1, pw_nm2, nm_b2, ne0, nullptr);
  // ee1 = MLP2(concat(x_i,x_j), em); msgs1 = mean_j(ee1)
  edge_encode_kernel<<<GNN_B * GNN_S, 256, 0, stream>>>(
      x, pw_em1, em_b1, pw_em2, em_b2, ee1, msgs1);
  // iter 1: ne1 = MLP2(concat(ne0, msgs1), nu)
  node_mlp_kernel<192, true, false><<<GNN_B, 512, 0, stream>>>(
      nullptr, ne0, msgs1, pw_nu1, nu_b1, pw_nu2, nu_b2, ne1, nullptr);
  //          ee2 = MLP2(concat(ee1, ne1_i, ne1_j), eu); msgs2 = mean_j(ee2)
  edge_update_kernel<true><<<GNN_B * GNN_S, 256, 0, stream>>>(
      ee1, ne1, pw_eu1, eu_b1, pw_eu2, eu_b2, ee2, msgs2, nullptr);
  // iter 2: ne2 = MLP2(concat(ne1, msgs2), nu); nsum = sum_s(ne2)
  node_mlp_kernel<192, true, true><<<GNN_B, 512, 0, stream>>>(
      nullptr, ne1, msgs2, pw_nu1, nu_b1, pw_nu2, nu_b2, ne2, nsum);
  //          final edge MLP: only accumulate esum = sum_{i,j}(ee3)
  edge_update_kernel<false><<<GNN_B * GNN_S, 256, 0, stream>>>(
      ee2, ne2, pw_eu1, eu_b1, pw_eu2, eu_b2, nullptr, nullptr, esum);
  // g = globals_mlp(concat(nsum/32, esum/1024)); out = (mu, softplus(logvar))
  globals_kernel<<<GNN_B, 256, 0, stream>>>(nsum, esum, gm_w1, gm_b1, gm_w2,
                                            gm_b2, gm_w3, gm_b3, (float*)d_out);
}
